// HSTUBlock_41308995453227
// MI455X (gfx1250) — compile-verified, matrix operand order checked by reading, not run on hardware
//
#include <hip/hip_runtime.h>
#include <hip/hip_bf16.h>
#include <math.h>

#define B_   8
#define N_   1024
#define H_   512
#define NH_  8
#define HD_  64
#define FF_  2048           // 4*H
#define TOK_ (B_ * N_)      // 8192

typedef __attribute__((ext_vector_type(16))) _Float16 v16h;
typedef __attribute__((ext_vector_type(8)))  _Float16 v8h;
typedef __attribute__((ext_vector_type(8)))  float    v8f;

// Per-lane fragment load for 16x32 f16 A-matrix (and the matching B layout):
// lane L -> row = L%16, K-chunk base = (L/16)*8 ; data = 8 f16 at +0 and 8 f16 at +16.
// Caller passes p = base + row*ld + kbase (16B aligned) -> two b128 loads.
__device__ __forceinline__ v16h load_frag(const _Float16* p) {
  v8h lo = *(const v8h*)(p);
  v8h hi = *(const v8h*)(p + 16);
  v16h r;
#pragma unroll
  for (int i = 0; i < 8; ++i) { r[i] = lo[i]; r[i + 8] = hi[i]; }
  return r;
}

__device__ __forceinline__ v8f wmma_f16(v16h a, v16h b, v8f c) {
  // (neg_a, A, neg_b, B, c_mod, C, reuse_a, reuse_b)
  return __builtin_amdgcn_wmma_f32_16x16x32_f16(false, a, false, b, (short)0, c,
                                                false, false);
}

__device__ __forceinline__ float silu_f(float v) {
  return v / (1.0f + __expf(-v));
}

// Software-pipelined 16x64 GEMM strip: acc[t] += A(16xK) * B_t(Kx16)^T over K=H_.
// aBase / bBase already include the per-lane row*ld + kb8 offset.
__device__ __forceinline__ void gemm_strip(const _Float16* __restrict__ aBase,
                                           const _Float16* __restrict__ bBase,
                                           v8f acc[4]) {
  v16h aC = load_frag(aBase);
  v16h bC0 = load_frag(bBase);
  v16h bC1 = load_frag(bBase + 16 * H_);
  v16h bC2 = load_frag(bBase + 32 * H_);
  v16h bC3 = load_frag(bBase + 48 * H_);
#pragma unroll
  for (int kk = 32; kk < H_; kk += 32) {
    // issue next-chunk loads before consuming current fragments
    v16h aN  = load_frag(aBase + kk);
    v16h bN0 = load_frag(bBase + kk);
    v16h bN1 = load_frag(bBase + 16 * H_ + kk);
    v16h bN2 = load_frag(bBase + 32 * H_ + kk);
    v16h bN3 = load_frag(bBase + 48 * H_ + kk);
    acc[0] = wmma_f16(aC, bC0, acc[0]);
    acc[1] = wmma_f16(aC, bC1, acc[1]);
    acc[2] = wmma_f16(aC, bC2, acc[2]);
    acc[3] = wmma_f16(aC, bC3, acc[3]);
    aC = aN; bC0 = bN0; bC1 = bN1; bC2 = bN2; bC3 = bN3;
  }
  acc[0] = wmma_f16(aC, bC0, acc[0]);
  acc[1] = wmma_f16(aC, bC1, acc[1]);
  acc[2] = wmma_f16(aC, bC2, acc[2]);
  acc[3] = wmma_f16(aC, bC3, acc[3]);
}

// ---------------------------------------------------------------- convert
__global__ void cvt_f32_f16(const float* __restrict__ src,
                            _Float16* __restrict__ dst, int n) {
  int i = blockIdx.x * blockDim.x + threadIdx.x;
  if (i < n) dst[i] = (_Float16)src[i];
}

// ---------------------------------------------------------------- bias
// bias[b,q,k] = pos_w[N-1+k-q] + ts_w[bucket(|ext[q+1]-ts[k]|)]
__global__ void bias_kernel(const int* __restrict__ ts_i,
                            const float* __restrict__ ts_w,
                            const float* __restrict__ pos_w,
                            float* __restrict__ bias) {
  unsigned idx = blockIdx.x * 256u + threadIdx.x;
  unsigned total = (unsigned)B_ * N_ * N_;
  if (idx >= total) return;
  int b = idx / (N_ * N_);
  int rem = idx % (N_ * N_);
  int q = rem / N_;
  int k = rem % N_;
  const int* tsb = ts_i + b * N_;
  float t_next = (float)tsb[(q + 1 < N_) ? (q + 1) : (N_ - 1)] * 1e-9f;
  float t_k    = (float)tsb[k] * 1e-9f;
  float diff = fabsf(t_next - t_k);
  diff = fmaxf(diff, 1.0f);
  int bucket = (int)floorf(__logf(diff) * (1.0f / 0.301f));
  bucket = min(max(bucket, 0), 64);
  bias[idx] = pos_w[N_ - 1 + k - q] + ts_w[bucket];
}

// ---------------------------------------------------------------- gemm1
// act = silu(x @ f1_w^T + f1_b); columns routed to U(f32) / Q(f16) / K(f16) / Vt(f16, d-major)
// grid (64, 32), block 256 (8 waves). Wave: 16 rows x 64 cols.
__global__ __launch_bounds__(256)
void gemm1_kernel(const _Float16* __restrict__ xh,
                  const _Float16* __restrict__ w1h,
                  const float* __restrict__ f1b,
                  float* __restrict__ U,
                  _Float16* __restrict__ Qh,
                  _Float16* __restrict__ Kh,
                  _Float16* __restrict__ Vt) {
  int wave = threadIdx.x >> 5;
  int lane = threadIdx.x & 31;
  int m0 = blockIdx.x * 128 + wave * 16;
  int n0 = blockIdx.y * 64;
  int row = lane & 15;
  int kb8 = (lane >> 4) << 3;

  v8f acc[4] = {};
  const _Float16* aBase = xh + (size_t)(m0 + row) * H_ + kb8;
  const _Float16* bBase = w1h + (size_t)(n0 + row) * H_ + kb8;
  gemm_strip(aBase, bBase, acc);

#pragma unroll
  for (int t = 0; t < 4; ++t) {
    int o = n0 + t * 16 + (lane & 15);
    float bv = f1b[o];
#pragma unroll
    for (int r = 0; r < 8; ++r) {
      int m = m0 + r + ((lane >> 4) << 3);
      float v = silu_f(acc[t][r] + bv);
      if (o < 512) {
        U[(size_t)m * H_ + o] = v;
      } else if (o < 1024) {
        Qh[(size_t)m * H_ + (o - 512)] = (_Float16)v;
      } else if (o < 1536) {
        Kh[(size_t)m * H_ + (o - 1024)] = (_Float16)v;
      } else {
        int c = o - 1536;
        int head = c >> 6, d = c & 63;
        int b = m >> 10, n = m & 1023;
        Vt[(((size_t)(b * NH_ + head)) * HD_ + d) * N_ + n] = (_Float16)v;
      }
    }
  }
}

// ---------------------------------------------------------------- attention
// Per wave: one (b, head, 16-row q tile); causal loop over k in 32-chunks.
// scores: A=Q frag (K=d), B=K frag; w re-layout via LDS; out: A=w frag (K=k), B=Vt frag.
__global__ __launch_bounds__(256)
void attn_kernel(const _Float16* __restrict__ Qh,
                 const _Float16* __restrict__ Kh,
                 const _Float16* __restrict__ Vt,
                 const float* __restrict__ bias,
                 const float* __restrict__ U,
                 _Float16* __restrict__ gatedh) {
  __shared__ __align__(16) _Float16 ldsw[8][16 * 32];
  int wave = threadIdx.x >> 5;
  int lane = threadIdx.x & 31;
  int task = blockIdx.x * 8 + wave;       // [0, 4096)
  int qt = task & 63;
  int head = (task >> 6) & 7;
  int b = task >> 9;
  int q0 = qt * 16;
  int row = lane & 15;
  int kb8 = (lane >> 4) << 3;

  const _Float16* qbase =
      Qh + (size_t)(b * N_ + q0 + row) * H_ + head * HD_ + kb8;
  v16h qf0 = load_frag(qbase);        // d = 0..31
  v16h qf1 = load_frag(qbase + 32);   // d = 32..63

  const _Float16* khead = Kh + (size_t)(b * N_) * H_ + head * HD_ + kb8;
  const _Float16* vhead = Vt + ((size_t)(b * NH_ + head)) * HD_ * N_;

  v8f out[4] = {};
  _Float16* myw = &ldsw[wave][0];

  for (int k0 = 0; k0 < q0 + 16; k0 += 32) {
    // hoist ALL loads of this k-chunk: K frags for both score tiles + V frags,
    // so V loads overlap the score WMMAs and the silu/bias epilogue.
    v16h kf00 = load_frag(khead + (size_t)(k0 + row) * H_);
    v16h kf01 = load_frag(khead + (size_t)(k0 + row) * H_ + 32);
    v16h kf10 = load_frag(khead + (size_t)(k0 + 16 + row) * H_);
    v16h kf11 = load_frag(khead + (size_t)(k0 + 16 + row) * H_ + 32);
    v16h vf[4];
#pragma unroll
    for (int d = 0; d < 4; ++d)
      vf[d] = load_frag(vhead + (size_t)(d * 16 + row) * N_ + k0 + kb8);
    if (k0 + 32 < q0 + 16)   // wave-uniform: prefetch next K chunk
      __builtin_prefetch(khead + (size_t)(k0 + 32 + row) * H_, 0, 0);

    v8f s0 = {};
    s0 = wmma_f16(qf0, kf00, s0);
    s0 = wmma_f16(qf1, kf01, s0);
    v8f s1 = {};
    s1 = wmma_f16(qf0, kf10, s1);
    s1 = wmma_f16(qf1, kf11, s1);

    int n = lane & 15;
#pragma unroll
    for (int t = 0; t < 2; ++t) {
      const v8f& s = t ? s1 : s0;
      int kg = k0 + t * 16 + n;
#pragma unroll
      for (int r = 0; r < 8; ++r) {
        int m = r + ((lane >> 4) << 3);
        int qg = q0 + m;
        float v = s[r] * 0.125f + bias[(size_t)(b * N_ + qg) * N_ + kg];
        v = silu_f(v);
        v = (kg <= qg) ? v : 0.0f;                 // causal mask
        myw[m * 32 + t * 16 + n] = (_Float16)v;    // re-layout via LDS
      }
    }
    v16h wa = load_frag(myw + row * 32 + kb8);     // 16x32 A fragment of w
#pragma unroll
    for (int d = 0; d < 4; ++d)
      out[d] = wmma_f16(wa, vf[d], out[d]);
  }

  // gated = attn_out * U, stored f16 for the final WMMA GEMM
#pragma unroll
  for (int d = 0; d < 4; ++d) {
    int col = head * HD_ + d * 16 + (lane & 15);
#pragma unroll
    for (int r = 0; r < 8; ++r) {
      int qg = q0 + r + ((lane >> 4) << 3);
      size_t off = (size_t)(b * N_ + qg) * H_ + col;
      gatedh[off] = (_Float16)(out[d][r] * U[off]);
    }
  }
}

// ---------------------------------------------------------------- gemm3
// out = gated @ f2_w^T + f2_b ; grid (64, 8), block 256.
__global__ __launch_bounds__(256)
void gemm3_kernel(const _Float16* __restrict__ gh,
                  const _Float16* __restrict__ w2h,
                  const float* __restrict__ f2b,
                  float* __restrict__ outp) {
  int wave = threadIdx.x >> 5;
  int lane = threadIdx.x & 31;
  int m0 = blockIdx.x * 128 + wave * 16;
  int n0 = blockIdx.y * 64;
  int row = lane & 15;
  int kb8 = (lane >> 4) << 3;

  v8f acc[4] = {};
  const _Float16* aBase = gh + (size_t)(m0 + row) * H_ + kb8;
  const _Float16* bBase = w2h + (size_t)(n0 + row) * H_ + kb8;
  gemm_strip(aBase, bBase, acc);

#pragma unroll
  for (int t = 0; t < 4; ++t) {
    int o = n0 + t * 16 + (lane & 15);
    float bv = f2b[o];
#pragma unroll
    for (int r = 0; r < 8; ++r) {
      int m = m0 + r + ((lane >> 4) << 3);
      outp[(size_t)m * H_ + o] = acc[t][r] + bv;
    }
  }
}

// ---------------------------------------------------------------- launch
extern "C" void kernel_launch(void* const* d_in, const int* in_sizes, int n_in,
                              void* d_out, int out_size, void* d_ws, size_t ws_size,
                              hipStream_t stream) {
  const float* x      = (const float*)d_in[0];
  const int*   ts     = (const int*)d_in[1];
  // d_in[2] attn_mask: causal, implemented analytically
  const float* f1_w   = (const float*)d_in[3];
  const float* f1_b   = (const float*)d_in[4];
  const float* f2_w   = (const float*)d_in[5];
  const float* f2_b   = (const float*)d_in[6];
  const float* ts_w   = (const float*)d_in[7];
  const float* pos_w  = (const float*)d_in[8];
  float* outp = (float*)d_out;

  char* w = (char*)d_ws;
  auto carve = [&](size_t bytes) {
    char* p = w;
    w += (bytes + 255) & ~(size_t)255;
    return p;
  };
  _Float16* xh    = (_Float16*)carve((size_t)TOK_ * H_ * 2);
  _Float16* w1h   = (_Float16*)carve((size_t)FF_ * H_ * 2);
  _Float16* w2h   = (_Float16*)carve((size_t)H_ * H_ * 2);
  float*    U     = (float*)   carve((size_t)TOK_ * H_ * 4);
  _Float16* Qh    = (_Float16*)carve((size_t)TOK_ * H_ * 2);
  _Float16* Kh    = (_Float16*)carve((size_t)TOK_ * H_ * 2);
  _Float16* Vt    = (_Float16*)carve((size_t)TOK_ * H_ * 2);
  _Float16* gh    = (_Float16*)carve((size_t)TOK_ * H_ * 2);
  float*    biasb = (float*)   carve((size_t)B_ * N_ * N_ * 4);

  int nx = TOK_ * H_;
  cvt_f32_f16<<<(nx + 255) / 256, 256, 0, stream>>>(x, xh, nx);
  int n1 = FF_ * H_;
  cvt_f32_f16<<<(n1 + 255) / 256, 256, 0, stream>>>(f1_w, w1h, n1);
  int n2 = H_ * H_;
  cvt_f32_f16<<<(n2 + 255) / 256, 256, 0, stream>>>(f2_w, w2h, n2);

  unsigned nb = (unsigned)B_ * N_ * N_;
  bias_kernel<<<(nb + 255) / 256, 256, 0, stream>>>(ts, ts_w, pos_w, biasb);

  gemm1_kernel<<<dim3(TOK_ / 128, FF_ / 64), 256, 0, stream>>>(
      xh, w1h, f1_b, U, Qh, Kh, Vt);

  attn_kernel<<<(B_ * NH_ * (N_ / 16)) / 8, 256, 0, stream>>>(
      Qh, Kh, Vt, biasb, U, gh);

  gemm3_kernel<<<dim3(TOK_ / 128, H_ / 64), 256, 0, stream>>>(
      gh, w2h, f2_b, outp);
}
